// AdjacencyModule_1657857376688
// MI455X (gfx1250) — matrix operand order repaired and checked by reading, not compiled, not run on hardware
//
#include <hip/hip_runtime.h>
#include <hip/hip_bf16.h>

#define B_ 4
#define N_ 512
#define D_ 128
#define H1_ 64
#define H2_ 32
#define NEG_SLOPE_ 0.1f

typedef __attribute__((ext_vector_type(16))) _Float16       v16h;
typedef __attribute__((ext_vector_type(8)))  _Float16       v8h;
typedef __attribute__((ext_vector_type(16))) unsigned short v16u;
typedef __attribute__((ext_vector_type(8)))  float          v8f;
typedef __attribute__((ext_vector_type(4)))  float          f32x4;

__device__ __forceinline__ float leaky(float x) {
    return x >= 0.0f ? x : NEG_SLOPE_ * x;
}

// |d| for a packed-f16 vector: clear sign bits.
__device__ __forceinline__ v16h habs16(v16h d) {
    v16u u = __builtin_bit_cast(v16u, d) & (unsigned short)0x7fff;
    return __builtin_bit_cast(v16h, u);
}

// One wave (wave32) owns one (b, i) row: computes scores for all 512 j's via
// WMMA (128->64->32) and does the softmax over j entirely within the wave.
__launch_bounds__(128, 1)
__global__ void adjacency_wmma_kernel(const float* __restrict__ v,
                                      const float* __restrict__ W1,
                                      const float* __restrict__ b1,
                                      const float* __restrict__ W2,
                                      const float* __restrict__ b2,
                                      const float* __restrict__ W3,
                                      const float* __restrict__ b3,
                                      float* __restrict__ out) {
    __shared__ __attribute__((aligned(32))) _Float16 lW1[H1_ * D_];   // [o][d] f16, 16 KB
    __shared__ __attribute__((aligned(32))) _Float16 lW2[H2_ * H1_];  // [o][c] f16, 4 KB
    __shared__ float lb1[H1_];
    __shared__ float lb2[H2_];
    __shared__ float lW3[H2_];
    __shared__ __attribute__((aligned(32))) _Float16 lh1[4][16 * H1_]; // per-wave h1 tile (16j x 64o)
    __shared__ float lscore[4][N_];                                    // per-wave score strip

    const int tid  = threadIdx.x;
    const int wave = tid >> 5;
    const int lane = tid & 31;
    const int ln16 = lane & 15;
    const int hi16 = lane >> 4;     // 0 or 1 (which half-wave)

    // ---- stage weights into LDS as f16 (cooperative, coalesced) ----
    for (int idx = tid; idx < H1_ * D_; idx += 128) lW1[idx] = (_Float16)W1[idx];
    for (int idx = tid; idx < H2_ * H1_; idx += 128) lW2[idx] = (_Float16)W2[idx];
    if (tid < H1_) lb1[tid] = b1[tid];
    if (tid < H2_) lb2[tid] = b2[tid];
    if (tid < H2_) lW3[tid] = W3[tid];
    __syncthreads();

    const int p = blockIdx.x * 4 + wave;   // flat (b,i) index, 2048 total
    const int b = p >> 9;                  // / N_
    const int i = p & (N_ - 1);
    const float b3v = b3[0];

    // ---- preload v_i slices, pre-converted to f16, in A-matrix lane layout ----
    // A-layout (16-bit, 16x32): lane l holds row M=l%16; its 16 halves are two
    // contiguous 8-element d-runs at d0 = 32c + 8*(l/16) and d0+16.
    const float* vrow_i = v + ((size_t)b * N_ + i) * D_;
    v16h vih[4];
    #pragma unroll
    for (int c = 0; c < 4; ++c) {
        const int d0 = 32 * c + 8 * hi16;
        f32x4 x0 = *(const f32x4*)(vrow_i + d0);
        f32x4 x1 = *(const f32x4*)(vrow_i + d0 + 4);
        f32x4 y0 = *(const f32x4*)(vrow_i + d0 + 16);
        f32x4 y1 = *(const f32x4*)(vrow_i + d0 + 20);
        #pragma unroll
        for (int e = 0; e < 4; ++e) {
            vih[c][e]      = (_Float16)x0[e];
            vih[c][e + 4]  = (_Float16)x1[e];
            vih[c][e + 8]  = (_Float16)y0[e];
            vih[c][e + 12] = (_Float16)y1[e];
        }
    }

    _Float16* myh1 = lh1[wave];
    float*    myS  = lscore[wave];

    // ==== loop over j-tiles of 16 ====
    for (int jt = 0; jt < N_ / 16; ++jt) {
        const int j = jt * 16 + ln16;     // this lane's j row
        const float* vrow_j = v + ((size_t)b * N_ + j) * D_;

        // prefetch next tile's v rows (global_prefetch_b8)
        if (jt + 1 < N_ / 16) {
            const float* nrow = vrow_j + 16 * D_;
            __builtin_prefetch(nrow, 0, 0);
            __builtin_prefetch(nrow + 32, 0, 0);
            __builtin_prefetch(nrow + 64, 0, 0);
            __builtin_prefetch(nrow + 96, 0, 0);
        }

        // ---- layer 1: (16j x 128) @ (128 x 64), bias in C ----
        v8f acc[4];
        #pragma unroll
        for (int ot = 0; ot < 4; ++ot) {
            const float bb = lb1[ot * 16 + ln16];
            #pragma unroll
            for (int r = 0; r < 8; ++r) acc[ot][r] = bb;
        }

        #pragma unroll
        for (int c = 0; c < 4; ++c) {
            // build A = |v_i - v_j| in packed f16 for this K=32 chunk
            const int d0 = 32 * c + 8 * hi16;
            f32x4 x0 = *(const f32x4*)(vrow_j + d0);
            f32x4 x1 = *(const f32x4*)(vrow_j + d0 + 4);
            f32x4 y0 = *(const f32x4*)(vrow_j + d0 + 16);
            f32x4 y1 = *(const f32x4*)(vrow_j + d0 + 20);
            v16h vjh;
            #pragma unroll
            for (int e = 0; e < 4; ++e) {
                vjh[e]      = (_Float16)x0[e];
                vjh[e + 4]  = (_Float16)x1[e];
                vjh[e + 8]  = (_Float16)y0[e];
                vjh[e + 12] = (_Float16)y1[e];
            }
            const v16h A = habs16(vjh - vih[c]);   // v_pk_add_f16(neg) + v_and

            #pragma unroll
            for (int ot = 0; ot < 4; ++ot) {
                // B-layout (32x16): lane l holds col N=l%16, K = 8*(l/16)+0..7 and 16+8*(l/16)+0..7
                // -> two 8-half runs of W1 row (o) at d=32c+8*(l/16) and d+16
                const _Float16* brow = &lW1[(ot * 16 + ln16) * D_ + 32 * c + 8 * hi16];
                const v8h b0 = *(const v8h*)(brow);
                const v8h b1v = *(const v8h*)(brow + 16);
                v16h Bm;
                #pragma unroll
                for (int e = 0; e < 8; ++e) { Bm[e] = b0[e]; Bm[e + 8] = b1v[e]; }
                acc[ot] = __builtin_amdgcn_wmma_f32_16x16x32_f16(
                    false, A, false, Bm, (short)0, acc[ot], false, false);
            }
        }

        // ---- leaky + stage h1 (16j x 64o) row-major f16 in this wave's LDS ----
        // C-layout: lane l, vgpr r -> (jloc = r + 8*(l/16), o = ot*16 + l%16)
        #pragma unroll
        for (int ot = 0; ot < 4; ++ot) {
            #pragma unroll
            for (int r = 0; r < 8; ++r) {
                myh1[(r + 8 * hi16) * H1_ + ot * 16 + ln16] = (_Float16)leaky(acc[ot][r]);
            }
        }
        // same-wave DS ops are in order: safe to read back without barrier

        // ---- layer 2: (16j x 64) @ (64 x 32) via 4 WMMAs ----
        v8f acc2[2];
        #pragma unroll
        for (int t2 = 0; t2 < 2; ++t2) {
            const float bb = lb2[t2 * 16 + ln16];
            #pragma unroll
            for (int r = 0; r < 8; ++r) acc2[t2][r] = bb;
        }
        #pragma unroll
        for (int c2 = 0; c2 < 2; ++c2) {
            const _Float16* arow = myh1 + ln16 * H1_ + 32 * c2 + 8 * hi16;
            v8h a0 = *(const v8h*)(arow);
            v8h a1 = *(const v8h*)(arow + 16);
            v16h A2;
            #pragma unroll
            for (int e = 0; e < 8; ++e) { A2[e] = a0[e]; A2[e + 8] = a1[e]; }
            #pragma unroll
            for (int t2 = 0; t2 < 2; ++t2) {
                const _Float16* b2row = &lW2[(t2 * 16 + ln16) * H1_ + 32 * c2 + 8 * hi16];
                const v8h q0 = *(const v8h*)(b2row);
                const v8h q1 = *(const v8h*)(b2row + 16);
                v16h B2;
                #pragma unroll
                for (int e = 0; e < 8; ++e) { B2[e] = q0[e]; B2[e + 8] = q1[e]; }
                acc2[t2] = __builtin_amdgcn_wmma_f32_16x16x32_f16(
                    false, A2, false, B2, (short)0, acc2[t2], false, false);
            }
        }

        // ---- layer 3 (32 -> 1): FMA + 16-lane butterfly straight from C-layout ----
        const float w30 = lW3[ln16];
        const float w31 = lW3[16 + ln16];
        #pragma unroll
        for (int r = 0; r < 8; ++r) {
            float psum = leaky(acc2[0][r]) * w30 + leaky(acc2[1][r]) * w31;
            psum += __shfl_xor(psum, 1, 32);
            psum += __shfl_xor(psum, 2, 32);
            psum += __shfl_xor(psum, 4, 32);
            psum += __shfl_xor(psum, 8, 32);   // reduced within each 16-lane half
            if (ln16 == 0) myS[jt * 16 + r + 8 * hi16] = psum + b3v;
        }
    }

    // ==== softmax over j (512 scores, all owned by this wave) ====
    float sv[16];
    float m = -3.402823466e38f;
    #pragma unroll
    for (int k = 0; k < 16; ++k) {
        sv[k] = myS[lane + 32 * k];
        m = fmaxf(m, sv[k]);
    }
    #pragma unroll
    for (int d = 1; d < 32; d <<= 1) m = fmaxf(m, __shfl_xor(m, d, 32));

    float ev[16];
    float s = 0.0f;
    #pragma unroll
    for (int k = 0; k < 16; ++k) {
        ev[k] = __expf(sv[k] - m);
        s += ev[k];
    }
    #pragma unroll
    for (int d = 1; d < 32; d <<= 1) s += __shfl_xor(s, d, 32);
    const float inv = 1.0f / s;

    float* orow = out + ((size_t)b * N_ + i) * N_;
    #pragma unroll
    for (int k = 0; k < 16; ++k) orow[lane + 32 * k] = ev[k] * inv;
}

extern "C" void kernel_launch(void* const* d_in, const int* in_sizes, int n_in,
                              void* d_out, int out_size, void* d_ws, size_t ws_size,
                              hipStream_t stream) {
    const float* v  = (const float*)d_in[0];
    const float* W1 = (const float*)d_in[1];
    const float* b1 = (const float*)d_in[2];
    const float* W2 = (const float*)d_in[3];
    const float* b2 = (const float*)d_in[4];
    const float* W3 = (const float*)d_in[5];
    const float* b3 = (const float*)d_in[6];
    float* out = (float*)d_out;

    dim3 grid(B_ * N_ / 4);   // 512 workgroups, 4 waves each -> 2048 (b,i) rows
    dim3 block(128);
    adjacency_wmma_kernel<<<grid, block, 0, stream>>>(v, W1, b1, W2, b2, W3, b3, out);
}
